// FastVARCrossAttnBlock_3925600109080
// MI455X (gfx1250) — matrix operand order repaired and checked by reading, not compiled, not run on hardware
//
#include <hip/hip_runtime.h>

typedef unsigned short u16;
typedef __bf16 bf16_t;
typedef __attribute__((ext_vector_type(16))) bf16_t v16bf;
typedef __attribute__((ext_vector_type(8)))  float  v8f;

#define WMMA_BF16(A_, B_, C_) \
  __builtin_amdgcn_wmma_f32_16x16x32_bf16(false, (A_), false, (B_), (short)0, (C_), false, false)

union AFrag { v16bf v; uint4 q[2]; };

// Problem constants (match reference setup).
constexpr int Bn  = 2;
constexpr int Ln  = 4096;   // 64*64
constexpr int Cn  = 1024;
constexpr int NHn = 16;
constexpr int DHn = 64;
constexpr int NRn = 1638;   // num_remain
constexpr int MPn = 1664;   // NRn padded up to multiple of 32 (52*32, 104*16)
constexpr int K3  = 3072;

__device__ __forceinline__ u16 f2bf(float f) {
  union { float f; unsigned int u; } v; v.f = f;
  unsigned int u = v.u;
  unsigned int r = u + 0x7FFFu + ((u >> 16) & 1u);  // round-to-nearest-even
  return (u16)(r >> 16);
}

// ---------------- stage 1: mean over L per (b,c) ----------------
__global__ void mean_kernel(const float* __restrict__ x, float* __restrict__ mean) {
  int t = blockIdx.x * blockDim.x + threadIdx.x;
  if (t >= Bn * Cn) return;
  int b = t / Cn, c = t % Cn;
  float s = 0.f;
  const float* p = x + (size_t)b * Ln * Cn + c;
  for (int l = 0; l < Ln; ++l) s += p[(size_t)l * Cn];
  mean[t] = s * (1.0f / Ln);
}

// ---------------- stage 2: mse per (b,l) ----------------
__global__ void mse_kernel(const float* __restrict__ x, const float* __restrict__ mean,
                           float* __restrict__ mse) {
  __shared__ float red[256];
  int blk = blockIdx.x;            // b*Ln + l
  int b = blk / Ln;
  int tid = threadIdx.x;
  const float* row = x + (size_t)blk * Cn;
  const float* mu  = mean + (size_t)b * Cn;
  float s = 0.f;
  for (int c = tid; c < Cn; c += 256) { float d = row[c] - mu[c]; s += d * d; }
  red[tid] = s; __syncthreads();
  for (int w = 128; w > 0; w >>= 1) { if (tid < w) red[tid] += red[tid + w]; __syncthreads(); }
  if (tid == 0) mse[blk] = red[0];
}

// ---------------- stage 3: stable descending rank -> top NR indices ----------------
__global__ void rank_kernel(const float* __restrict__ mse, int* __restrict__ idx) {
  __shared__ int red[256];
  int blk = blockIdx.x;            // b*Ln + l
  int b = blk / Ln, l = blk % Ln;
  int tid = threadIdx.x;
  const float* m = mse + (size_t)b * Ln;
  float mv = m[l];
  int cnt = 0;
  for (int j = tid; j < Ln; j += 256) {
    float mj = m[j];
    if (mj > mv || (mj == mv && j < l)) cnt++;
  }
  red[tid] = cnt; __syncthreads();
  for (int w = 128; w > 0; w >>= 1) { if (tid < w) red[tid] += red[tid + w]; __syncthreads(); }
  if (tid == 0) {
    int rank = red[0];
    if (rank < NRn) idx[b * MPn + rank] = l;
  }
}

// ---------------- fp32 -> bf16 convert ----------------
__global__ void cvt_kernel(const float* __restrict__ src, u16* __restrict__ dst, int n) {
  for (int i = blockIdx.x * blockDim.x + threadIdx.x; i < n; i += gridDim.x * blockDim.x)
    dst[i] = f2bf(src[i]);
}

// ---------------- build concat(q_bias, 0, v_bias) ----------------
__global__ void bias_kernel(const float* __restrict__ qb, const float* __restrict__ vb,
                            float* __restrict__ bias3) {
  int t = blockIdx.x * blockDim.x + threadIdx.x;
  if (t >= K3) return;
  bias3[t] = (t < Cn) ? qb[t] : (t < 2 * Cn) ? 0.f : vb[t - 2 * Cn];
}

// ---------------- gather selected rows of x into bf16 (zero-padded) ----------------
__global__ void gather_kernel(const float* __restrict__ x, const int* __restrict__ idx,
                              u16* __restrict__ xm) {
  int r = blockIdx.x, b = blockIdx.y;
  u16* drow = xm + ((size_t)b * MPn + r) * Cn;
  if (r < NRn) {
    int l = idx[b * MPn + r];
    const float* srow = x + ((size_t)b * Ln + l) * Cn;
    for (int c = threadIdx.x; c < Cn; c += 256) drow[c] = f2bf(srow[c]);
  } else {
    for (int c = threadIdx.x; c < Cn; c += 256) drow[c] = 0;
  }
}

// ---------------- bf16 WMMA GEMM: D[M,N] = A[M,K] * B[N,K]^T + bias[N] ----------------
// block = (32,4): 4 waves, each wave does a 16x64 output tile. Branch-free k-loop
// so the compiler software-pipelines loads against WMMAs (s_wait_loadcnt partial waits).
__global__ void gemm_bf16_kernel(const u16* __restrict__ A, const u16* __restrict__ B,
                                 const float* __restrict__ bias, float* __restrict__ D,
                                 int M, int N, int K) {
  int lane = threadIdx.x;
  int n0 = (blockIdx.x * 4 + threadIdx.y) * 64;
  int m0 = blockIdx.y * 16;
  if (n0 >= N || m0 >= M) return;
  int nc = lane & 15, half = lane >> 4;
  int bka = half ? 16 : 0;    // A frag K base
  int bkb = half ? 16 : 0;    // B frag K base
  const u16* arow = A + (size_t)(m0 + nc) * K;
  v8f a0 = {}, a1 = {}, a2 = {}, a3 = {};
  for (int k0 = 0; k0 < K; k0 += 32) {
    AFrag af;
    af.q[0] = *(const uint4*)(arow + k0 + bka);
    af.q[1] = *(const uint4*)(arow + k0 + bka + 8);
    const u16* bbase = B + (size_t)(n0 + nc) * K + k0 + bkb;
    v16bf b0 = *(const v16bf*)(bbase);
    v16bf b1 = *(const v16bf*)(bbase + (size_t)16 * K);
    v16bf b2 = *(const v16bf*)(bbase + (size_t)32 * K);
    v16bf b3 = *(const v16bf*)(bbase + (size_t)48 * K);
    a0 = WMMA_BF16(af.v, b0, a0);
    a1 = WMMA_BF16(af.v, b1, a1);
    a2 = WMMA_BF16(af.v, b2, a2);
    a3 = WMMA_BF16(af.v, b3, a3);
  }
  int rbase = half ? 8 : 0;
#pragma unroll
  for (int t = 0; t < 4; ++t) {
    int col = n0 + t * 16 + nc;
    float bv = bias[col];
    v8f acc = (t == 0) ? a0 : (t == 1) ? a1 : (t == 2) ? a2 : a3;
#pragma unroll
    for (int r = 0; r < 8; ++r)
      D[(size_t)(m0 + rbase + r) * N + col] = acc[r] + bv;
  }
}

// ---------------- l2-norm + scale + RoPE, emit bf16 Q,K and V^T ----------------
__global__ void qk_prep_kernel(const float* __restrict__ qkv, const int* __restrict__ idx,
                               const float* __restrict__ smlog, const float* __restrict__ rope,
                               u16* __restrict__ Qb, u16* __restrict__ Kb, u16* __restrict__ Vt) {
  int m = blockIdx.x, h = blockIdx.y, b = blockIdx.z;
  int d = threadIdx.x;
  __shared__ float sq[64], sk[64];
  size_t qkoff = ((size_t)(b * NHn + h) * MPn + m) * DHn;
  size_t voff  = ((size_t)(b * NHn + h) * DHn + d) * MPn + m;
  if (m >= NRn) { Qb[qkoff + d] = 0; Kb[qkoff + d] = 0; Vt[voff] = 0; return; }
  size_t row = (size_t)(b * MPn + m) * K3;
  float qv = qkv[row + h * DHn + d];
  float kv = qkv[row + Cn + h * DHn + d];
  float vv = qkv[row + 2 * Cn + h * DHn + d];
  sq[d] = qv * qv; sk[d] = kv * kv;
  __syncthreads();
  for (int s = 32; s > 0; s >>= 1) {
    if (d < s) { sq[d] += sq[d + s]; sk[d] += sk[d + s]; }
    __syncthreads();
  }
  float invq = 1.0f / fmaxf(sqrtf(sq[0]), 1e-12f);
  float invk = 1.0f / fmaxf(sqrtf(sk[0]), 1e-12f);
  float sm = __expf(fminf(smlog[h], 4.6051702f));  // min(log, log(100))
  int l = idx[b * MPn + m];
  int j = d >> 1;
  float r0 = rope[(size_t)l * 32 + j];
  float r1 = rope[(size_t)Ln * 32 + (size_t)l * 32 + j];
  float q0 = qkv[row + h * DHn + (j << 1)]     * invq * sm;
  float q1 = qkv[row + h * DHn + (j << 1) + 1] * invq * sm;
  float k0 = qkv[row + Cn + h * DHn + (j << 1)]     * invk;
  float k1 = qkv[row + Cn + h * DHn + (j << 1) + 1] * invk;
  float qr = (d & 1) ? (r1 * q0 + r0 * q1) : (r0 * q0 - r1 * q1);
  float kr = (d & 1) ? (r1 * k0 + r0 * k1) : (r0 * k0 - r1 * k1);
  Qb[qkoff + d] = f2bf(qr);
  Kb[qkoff + d] = f2bf(kr);
  Vt[voff] = f2bf(vv);
}

// ---------------- flash attention: one wave per (b,h,16-query tile) ----------------
__global__ void attn_kernel(const u16* __restrict__ Qb, const u16* __restrict__ Kb,
                            const u16* __restrict__ Vt, u16* __restrict__ Oat) {
  __shared__ u16 pls[16 * 32];   // P tile, row-major 16x32
  int lane = threadIdx.x;
  int m0 = blockIdx.x * 16, h = blockIdx.y, b = blockIdx.z;
  int nc = lane & 15, half = lane >> 4;
  int bka = half ? 16 : 0, bkb = half ? 16 : 0;
  const u16* Q  = Qb + (size_t)(b * NHn + h) * MPn * DHn;
  const u16* Km = Kb + (size_t)(b * NHn + h) * MPn * DHn;
  const u16* V  = Vt + (size_t)(b * NHn + h) * DHn * MPn;
  AFrag qa0, qa1;
  const u16* qrow = Q + (size_t)(m0 + nc) * DHn;
  qa0.q[0] = *(const uint4*)(qrow + bka);
  qa0.q[1] = *(const uint4*)(qrow + bka + 8);
  qa1.q[0] = *(const uint4*)(qrow + 32 + bka);
  qa1.q[1] = *(const uint4*)(qrow + 32 + bka + 8);
  v8f o0 = {}, o1 = {}, o2 = {}, o3 = {};
  float rm[8], rl[8], rsc[8];
#pragma unroll
  for (int r = 0; r < 8; ++r) { rm[r] = -1e30f; rl[r] = 0.f; }

  for (int kb0 = 0; kb0 < MPn; kb0 += 32) {
    v8f s0 = {}, s1 = {};
    {
      const u16* k0p = Km + (size_t)(kb0 + nc) * DHn + bkb;
      const u16* k1p = Km + (size_t)(kb0 + 16 + nc) * DHn + bkb;
      v16bf kf;
      kf = *(const v16bf*)(k0p);      s0 = WMMA_BF16(qa0.v, kf, s0);
      kf = *(const v16bf*)(k0p + 32); s0 = WMMA_BF16(qa1.v, kf, s0);
      kf = *(const v16bf*)(k1p);      s1 = WMMA_BF16(qa0.v, kf, s1);
      kf = *(const v16bf*)(k1p + 32); s1 = WMMA_BF16(qa1.v, kf, s1);
    }
    bool ok0 = (kb0 + nc) < NRn, ok1 = (kb0 + 16 + nc) < NRn;
#pragma unroll
    for (int r = 0; r < 8; ++r) {
      float v0 = ok0 ? s0[r] : -1e30f;
      float v1 = ok1 ? s1[r] : -1e30f;
      float mx = fmaxf(v0, v1);
      mx = fmaxf(mx, __shfl_xor(mx, 1, 32));
      mx = fmaxf(mx, __shfl_xor(mx, 2, 32));
      mx = fmaxf(mx, __shfl_xor(mx, 4, 32));
      mx = fmaxf(mx, __shfl_xor(mx, 8, 32));
      float mnew = fmaxf(rm[r], mx);
      float sc = __expf(rm[r] - mnew);
      float p0 = ok0 ? __expf(v0 - mnew) : 0.f;
      float p1 = ok1 ? __expf(v1 - mnew) : 0.f;
      float rs = p0 + p1;
      rs += __shfl_xor(rs, 1, 32);
      rs += __shfl_xor(rs, 2, 32);
      rs += __shfl_xor(rs, 4, 32);
      rs += __shfl_xor(rs, 8, 32);
      rl[r] = rl[r] * sc + rs;
      rm[r] = mnew;
      rsc[r] = sc;
      int prow = half ? (8 + r) : r;
      pls[prow * 32 + nc]      = f2bf(p0);
      pls[prow * 32 + 16 + nc] = f2bf(p1);
    }
#pragma unroll
    for (int r = 0; r < 8; ++r) {
      o0[r] *= rsc[r]; o1[r] *= rsc[r]; o2[r] *= rsc[r]; o3[r] *= rsc[r];
    }
    __syncthreads();
    AFrag pa;  // P as A-fragment: row = nc for both halves
    pa.q[0] = *(const uint4*)(&pls[nc * 32 + bka]);
    pa.q[1] = *(const uint4*)(&pls[nc * 32 + bka + 8]);
    const u16* vcol = V + (size_t)nc * MPn + kb0 + bkb;
    v16bf vf;
    vf = *(const v16bf*)(vcol);                    o0 = WMMA_BF16(pa.v, vf, o0);
    vf = *(const v16bf*)(vcol + (size_t)16 * MPn); o1 = WMMA_BF16(pa.v, vf, o1);
    vf = *(const v16bf*)(vcol + (size_t)32 * MPn); o2 = WMMA_BF16(pa.v, vf, o2);
    vf = *(const v16bf*)(vcol + (size_t)48 * MPn); o3 = WMMA_BF16(pa.v, vf, o3);
    __syncthreads();
  }
  int rbase = half ? 8 : 0;
#pragma unroll
  for (int r = 0; r < 8; ++r) {
    float inv = 1.0f / rl[r];
    size_t rowoff = (size_t)(b * MPn + m0 + rbase + r) * Cn + (size_t)h * DHn;
    Oat[rowoff + nc]      = f2bf(o0[r] * inv);
    Oat[rowoff + 16 + nc] = f2bf(o1[r] * inv);
    Oat[rowoff + 32 + nc] = f2bf(o2[r] * inv);
    Oat[rowoff + 48 + nc] = f2bf(o3[r] * inv);
  }
}

// ---------------- out = x + upsample2x(cached_x) everywhere ----------------
__global__ void final_base_kernel(const float* __restrict__ x, const float* __restrict__ cached,
                                  float* __restrict__ out) {
  size_t total = (size_t)Bn * Ln * Cn;
  for (size_t i = (size_t)blockIdx.x * blockDim.x + threadIdx.x; i < total;
       i += (size_t)gridDim.x * blockDim.x) {
    size_t b = i / ((size_t)Ln * Cn);
    size_t rem = i % ((size_t)Ln * Cn);
    int l = (int)(rem / Cn), c = (int)(rem % Cn);
    int hh = l >> 6, ww = l & 63;
    float up = cached[(((size_t)b * 32 + (hh >> 1)) * 32 + (ww >> 1)) * Cn + c];
    out[i] = x[i] + up;
  }
}

// ---------------- scatter attention output rows: out[idx] = x[idx] + o_proj ----------------
__global__ void final_scatter_kernel(const float* __restrict__ x, const int* __restrict__ idx,
                                     const float* __restrict__ oproj, float* __restrict__ out) {
  int r = blockIdx.x, b = blockIdx.y;
  int l = idx[b * MPn + r];
  size_t dst = ((size_t)b * Ln + l) * Cn;
  size_t src = ((size_t)b * MPn + r) * Cn;
  for (int c = threadIdx.x; c < Cn; c += 256)
    out[dst + c] = x[dst + c] + oproj[src + c];
}

extern "C" void kernel_launch(void* const* d_in, const int* in_sizes, int n_in,
                              void* d_out, int out_size, void* d_ws, size_t ws_size,
                              hipStream_t stream) {
  const float* x      = (const float*)d_in[0];
  const float* cached = (const float*)d_in[1];
  const float* Wqkv   = (const float*)d_in[2];
  const float* qbias  = (const float*)d_in[3];
  const float* vbias  = (const float*)d_in[4];
  const float* Wproj  = (const float*)d_in[5];
  const float* bproj  = (const float*)d_in[6];
  const float* smlog  = (const float*)d_in[7];
  const float* rope   = (const float*)d_in[8];
  float* out = (float*)d_out;

  char* p = (char*)d_ws;
  auto alloc = [&](size_t bytes) -> char* {
    char* r = p; p += (bytes + 255) & ~(size_t)255; return r;
  };
  const int Mtot = Bn * MPn;  // 3328
  float* mean   = (float*)alloc((size_t)Bn * Cn * 4);
  float* mse    = (float*)alloc((size_t)Bn * Ln * 4);
  int*   idx    = (int*)alloc((size_t)Bn * MPn * 4);
  float* bias3  = (float*)alloc((size_t)K3 * 4);
  u16*   WqkvB  = (u16*)alloc((size_t)K3 * Cn * 2);
  u16*   WprojB = (u16*)alloc((size_t)Cn * Cn * 2);
  u16*   xmB    = (u16*)alloc((size_t)Mtot * Cn * 2);
  float* qkvF   = (float*)alloc((size_t)Mtot * K3 * 4);
  u16*   Qb     = (u16*)alloc((size_t)Bn * NHn * MPn * DHn * 2);
  u16*   Kb     = (u16*)alloc((size_t)Bn * NHn * MPn * DHn * 2);
  u16*   Vt     = (u16*)alloc((size_t)Bn * NHn * DHn * MPn * 2);
  u16*   OatB   = (u16*)alloc((size_t)Mtot * Cn * 2);
  // qkvF (40.9 MB) is dead after qk_prep; alias the proj output (13.6 MB) into it.
  float* OprojF = qkvF;

  // 1-3: token selection
  mean_kernel<<<(Bn * Cn + 255) / 256, 256, 0, stream>>>(x, mean);
  mse_kernel<<<Bn * Ln, 256, 0, stream>>>(x, mean, mse);
  rank_kernel<<<Bn * Ln, 256, 0, stream>>>(mse, idx);

  // weight / bias prep
  cvt_kernel<<<2048, 256, 0, stream>>>(Wqkv, WqkvB, K3 * Cn);
  cvt_kernel<<<1024, 256, 0, stream>>>(Wproj, WprojB, Cn * Cn);
  bias_kernel<<<(K3 + 255) / 256, 256, 0, stream>>>(qbias, vbias, bias3);

  // gather + QKV GEMM
  gather_kernel<<<dim3(MPn, Bn), 256, 0, stream>>>(x, idx, xmB);
  gemm_bf16_kernel<<<dim3(K3 / 256, Mtot / 16), dim3(32, 4), 0, stream>>>(
      xmB, WqkvB, bias3, qkvF, Mtot, K3, Cn);

  // norm + rope + layout
  qk_prep_kernel<<<dim3(MPn, NHn, Bn), 64, 0, stream>>>(qkvF, idx, smlog, rope, Qb, Kb, Vt);

  // flash attention
  attn_kernel<<<dim3(MPn / 16, NHn, Bn), 32, 0, stream>>>(Qb, Kb, Vt, OatB);

  // output projection (writes into aliased qkvF space; qkvF no longer needed)
  gemm_bf16_kernel<<<dim3(Cn / 256, Mtot / 16), dim3(32, 4), 0, stream>>>(
      OatB, WprojB, bproj, OprojF, Mtot, Cn, Cn);

  // residual + upsample + scatter
  final_base_kernel<<<32768, 256, 0, stream>>>(x, cached, out);
  final_scatter_kernel<<<dim3(NRn, Bn), 256, 0, stream>>>(x, idx, OprojF, out);
}
